// MultiHeadGAT_5738076307731
// MI455X (gfx1250) — matrix operand order repaired and checked by reading, not compiled, not run on hardware
//
#include <hip/hip_runtime.h>
#include <hip/hip_bf16.h>

// Problem constants (from reference)
#define NN        20000
#define NE        320000
#define IND       128
#define HIDD      128
#define NHEADS    4
#define NB        16
#define NACT      10
#define NEG_SLOPE 0.2f

typedef __attribute__((ext_vector_type(16))) __bf16 v16bf;
typedef __attribute__((ext_vector_type(8)))  __bf16 v8bf;
typedef __attribute__((ext_vector_type(8)))  float  v8f;

// Monotonic float <-> uint mapping so atomicMax(uint) implements float max.
__device__ __forceinline__ unsigned fmap(float f) {
  unsigned u = __float_as_uint(f);
  return (u & 0x80000000u) ? ~u : (u | 0x80000000u);
}
__device__ __forceinline__ float funmap(unsigned u) {
  return (u & 0x80000000u) ? __uint_as_float(u & 0x7fffffffu) : __uint_as_float(~u);
}
__device__ __forceinline__ float elu1(float v) { return v > 0.f ? v : (__expf(v) - 1.f); }

// ---------------------------------------------------------------------------
// K0: pack a K=128 x Nc f32 weight matrix into WMMA B-fragment order (bf16).
// out flat index = ((tile*4 + kblk)*32 + lane)*16 + e, where
//   K = kblk*32 + (e&7) + 8*(lane>=16) + 16*(e>=8),  N = tile*16 + (lane&15).
// After packing, a wave's B fragment is ONE aligned 32-byte v16bf load.
// ---------------------------------------------------------------------------
__global__ void k_pack_w(const float* __restrict__ W, int Nc,
                         __bf16* __restrict__ out, int total) {
  int idx = blockIdx.x * blockDim.x + threadIdx.x;
  if (idx >= total) return;                  // total = 128 * Nc
  int e    = idx & 15;
  int lane = (idx >> 4) & 31;
  int kb   = (idx >> 9) & 3;
  int t    = idx >> 11;
  int k = kb * 32 + (e & 7) + 8 * (lane >> 4) + 16 * (e >> 3);
  int n = t * 16 + (lane & 15);
  out[idx] = (__bf16)W[k * Nc + n];
}

// ---------------------------------------------------------------------------
// K1: SAGE neighbor scatter: neigh[dst] += x[src]; deg[dst] += 1
// One wave32 per edge, each lane moves a float4 (128 floats/row).
// ---------------------------------------------------------------------------
__global__ void k_sage_scatter(const float* __restrict__ x, const int* __restrict__ ei,
                               float* __restrict__ neigh, float* __restrict__ deg) {
  long tid = (long)blockIdx.x * blockDim.x + threadIdx.x;
  int e = (int)(tid >> 5);
  if (e >= NE) return;
  int lane = (int)(tid & 31);
  int src = ei[e], dst = ei[NE + e];
  float4 v = ((const float4*)(x + (long)src * IND))[lane];
  float* nd = neigh + (long)dst * IND + lane * 4;
  atomicAdd(nd + 0, v.x);
  atomicAdd(nd + 1, v.y);
  atomicAdd(nd + 2, v.z);
  atomicAdd(nd + 3, v.w);
  if (lane == 0) atomicAdd(deg + dst, 1.0f);
}

// ---------------------------------------------------------------------------
// K1b: one-pass f32->bf16 conversion of the two A operands:
//   x_bf = bf16(x),  mean_bf = bf16(neigh_sum / max(deg,1))
// Hoists all conversion VALU out of the GEMM (was done 8x per row there).
// ---------------------------------------------------------------------------
__global__ void k_cvt_inputs(const float* __restrict__ x, const float* __restrict__ neigh,
                             const float* __restrict__ deg,
                             __bf16* __restrict__ xbf, __bf16* __restrict__ meanbf) {
  int t = blockIdx.x * blockDim.x + threadIdx.x;
  if (t >= NN * HIDD) return;
  int n = t >> 7;
  float dinv = 1.0f / fmaxf(deg[n], 1.0f);
  xbf[t]    = (__bf16)x[t];
  meanbf[t] = (__bf16)(neigh[t] * dinv);
}

// ---------------------------------------------------------------------------
// Helper: build a 16-bit WMMA A fragment from a bf16 row.
// Per ISA layout: K = (e&7) + 8*half + 16*(e>=8) -> two aligned 16B runs.
// ---------------------------------------------------------------------------
__device__ __forceinline__ v16bf load_a_frag(const __bf16* __restrict__ row,
                                             int K0, int gg) {
  v8bf lo = *(const v8bf*)(row + K0 + 8 * gg);
  v8bf hi = *(const v8bf*)(row + K0 + 8 * gg + 16);
  v16bf a;
#pragma unroll
  for (int e = 0; e < 8; ++e) { a[e] = lo[e]; a[e + 8] = hi[e]; }
  return a;
}

// ---------------------------------------------------------------------------
// K2: h = elu(mean @ W_l + x @ W_r + b)  via WMMA bf16; h stored bf16.
// Block = 16 rows, 8 waves; wave w owns columns [16w, 16w+16).
// All fragments are aligned vector loads; no conversions in the loop.
// ---------------------------------------------------------------------------
__global__ void __launch_bounds__(256) k_sage_gemm(
    const __bf16* __restrict__ xbf, const __bf16* __restrict__ meanbf,
    const __bf16* __restrict__ Wlp, const __bf16* __restrict__ Wrp,
    const float* __restrict__ bs, __bf16* __restrict__ hbf) {
  const int lane  = threadIdx.x & 31;
  const int wave  = threadIdx.x >> 5;
  const int gg    = lane >> 4;     // lane half selects K sub-range
  const int ln    = lane & 15;
  const int mbase = blockIdx.x * 16;
  const int nbase = wave * 16;
  const __bf16* mrow = meanbf + (long)(mbase + ln) * HIDD;
  const __bf16* xrow = xbf + (long)(mbase + ln) * HIDD;
  const v16bf* BL = (const v16bf*)Wlp;
  const v16bf* BR = (const v16bf*)Wrp;

  v8f c = {};
#pragma unroll
  for (int kb = 0; kb < 4; ++kb) {
    const int K0 = kb * 32;
    v16bf aN = load_a_frag(mrow, K0, gg);
    v16bf aX = load_a_frag(xrow, K0, gg);
    v16bf bL = BL[(wave * 4 + kb) * 32 + lane];
    v16bf bR = BR[(wave * 4 + kb) * 32 + lane];
    c = __builtin_amdgcn_wmma_f32_16x16x32_bf16(false, aN, false, bL, (short)0, c, false, false);
    c = __builtin_amdgcn_wmma_f32_16x16x32_bf16(false, aX, false, bR, (short)0, c, false, false);
  }
  const float bias = bs[nbase + ln];
#pragma unroll
  for (int r = 0; r < 8; ++r) {
    int m = mbase + r + 8 * gg;        // C/D layout: M = r + 8*half, N = lane&15
    hbf[(long)m * HIDD + nbase + ln] = (__bf16)elu1(c[r] + bias);
  }
}

// ---------------------------------------------------------------------------
// K3: g = h @ W_gat  ([N,128] x [128,512]) via WMMA bf16, h already bf16.
// grid = (N/16, 4); wave tile column base = (by*8+wave)*16.
// ---------------------------------------------------------------------------
__global__ void __launch_bounds__(256) k_gat_gemm(
    const __bf16* __restrict__ hbf, const __bf16* __restrict__ Wgp,
    float* __restrict__ gbuf) {
  const int lane  = threadIdx.x & 31;
  const int wave  = threadIdx.x >> 5;
  const int gg    = lane >> 4;
  const int ln    = lane & 15;
  const int mbase = blockIdx.x * 16;
  const int tg    = blockIdx.y * 8 + wave;           // column tile 0..31
  const int nbase = tg * 16;                         // 0..496
  const __bf16* hrow = hbf + (long)(mbase + ln) * HIDD;
  const v16bf* BG = (const v16bf*)Wgp;

  v8f c = {};
#pragma unroll
  for (int kb = 0; kb < 4; ++kb) {
    v16bf a = load_a_frag(hrow, kb * 32, gg);
    v16bf b = BG[(tg * 4 + kb) * 32 + lane];
    c = __builtin_amdgcn_wmma_f32_16x16x32_bf16(false, a, false, b, (short)0, c, false, false);
  }
#pragma unroll
  for (int r = 0; r < 8; ++r) {
    int m = mbase + r + 8 * gg;
    gbuf[(long)m * (NHEADS * HIDD) + nbase + ln] = c[r];
  }
}

// ---------------------------------------------------------------------------
// K4: per-node per-head attention scores a_s, a_d (dot over 128).
// ---------------------------------------------------------------------------
__global__ void k_att_scores(const float* __restrict__ gbuf,
                             const float* __restrict__ att_s, const float* __restrict__ att_d,
                             float* __restrict__ a_s, float* __restrict__ a_d) {
  int t = blockIdx.x * blockDim.x + threadIdx.x;
  if (t >= NN * NHEADS) return;
  int n = t >> 2, hh = t & 3;
  const float* gp = gbuf + (long)n * (NHEADS * HIDD) + hh * HIDD;
  const float* ws = att_s + hh * HIDD;
  const float* wd = att_d + hh * HIDD;
  float ss = 0.f, sd = 0.f;
#pragma unroll 4
  for (int c = 0; c < HIDD; ++c) { float gv = gp[c]; ss += gv * ws[c]; sd += gv * wd[c]; }
  a_s[t] = ss;
  a_d[t] = sd;
}

// ---------------------------------------------------------------------------
// K5: e = leaky_relu(a_s[src]+a_d[dst]); segment max via uint atomicMax.
// ---------------------------------------------------------------------------
__global__ void k_edge_max(const int* __restrict__ ei, const float* __restrict__ a_s,
                           const float* __restrict__ a_d, float* __restrict__ ebuf,
                           unsigned* __restrict__ mmax) {
  int t = blockIdx.x * blockDim.x + threadIdx.x;
  if (t >= NE * NHEADS) return;
  int e = t >> 2, hh = t & 3;
  int src = ei[e], dst = ei[NE + e];
  float v = a_s[src * NHEADS + hh] + a_d[dst * NHEADS + hh];
  v = v > 0.f ? v : NEG_SLOPE * v;
  ebuf[t] = v;
  atomicMax(mmax + dst * NHEADS + hh, fmap(v));
}

// ---------------------------------------------------------------------------
// K6: ex = exp(e - max[dst]); denom[dst] += ex  (ebuf overwritten with ex)
// ---------------------------------------------------------------------------
__global__ void k_edge_exp(const int* __restrict__ ei, const unsigned* __restrict__ mmax,
                           float* __restrict__ ebuf, float* __restrict__ denom) {
  int t = blockIdx.x * blockDim.x + threadIdx.x;
  if (t >= NE * NHEADS) return;
  int e = t >> 2, hh = t & 3;
  int dst = ei[NE + e];
  float ex = __expf(ebuf[t] - funmap(mmax[dst * NHEADS + hh]));
  ebuf[t] = ex;
  atomicAdd(denom + dst * NHEADS + hh, ex);
}

// ---------------------------------------------------------------------------
// K7: h2[dst] += (1/H) * sum_h alpha[e,h] * g[src,h,:]   (wave per edge)
// ---------------------------------------------------------------------------
__global__ void k_edge_aggr(const int* __restrict__ ei, const float* __restrict__ ebuf,
                            const float* __restrict__ denom, const float* __restrict__ gbuf,
                            float* __restrict__ h2) {
  long tid = (long)blockIdx.x * blockDim.x + threadIdx.x;
  int e = (int)(tid >> 5);
  if (e >= NE) return;
  int lane = (int)(tid & 31);
  int src = ei[e], dst = ei[NE + e];
  float alpha[NHEADS];
#pragma unroll
  for (int hh = 0; hh < NHEADS; ++hh)
    alpha[hh] = 0.25f * ebuf[e * NHEADS + hh] / (denom[dst * NHEADS + hh] + 1e-16f);
  const float* gp = gbuf + (long)src * (NHEADS * HIDD);
  float* op = h2 + (long)dst * HIDD;
#pragma unroll
  for (int j = 0; j < 4; ++j) {
    int cidx = lane + 32 * j;
    float acc = 0.f;
#pragma unroll
    for (int hh = 0; hh < NHEADS; ++hh) acc += alpha[hh] * gp[hh * HIDD + cidx];
    atomicAdd(op + cidx, acc);
  }
}

// ---------------------------------------------------------------------------
// K8: h2 = elu(h2 + b_gat); accumulate pooled sums and counts per batch.
// ---------------------------------------------------------------------------
__global__ void k_finalize(const float* __restrict__ bg, const int* __restrict__ batch,
                           float* __restrict__ h2, float* __restrict__ pooled,
                           float* __restrict__ counts) {
  int t = blockIdx.x * blockDim.x + threadIdx.x;
  if (t >= NN * HIDD) return;
  int n = t >> 7, cc = t & 127;
  float v = elu1(h2[t] + bg[cc]);
  h2[t] = v;
  int b = batch[n];
  atomicAdd(pooled + b * HIDD + cc, v);
  if (cc == 0) atomicAdd(counts + b, 1.0f);
}

// ---------------------------------------------------------------------------
// K9: wire / term sigmoid heads (one thread per node, dot over 128).
// out[0 .. N-11] = wire_probs, out[N-10 .. N-1] = term_probs.
// ---------------------------------------------------------------------------
__global__ void k_node_heads(const float* __restrict__ h2,
                             const float* __restrict__ Ww, const float* __restrict__ bw,
                             const float* __restrict__ Wt, const float* __restrict__ bt,
                             float* __restrict__ out) {
  int n = blockIdx.x * blockDim.x + threadIdx.x;
  if (n >= NN) return;
  bool isTerm = (n >= NN - 10);
  const float* W = isTerm ? Wt : Ww;
  float s = isTerm ? bt[0] : bw[0];
  const float* hp = h2 + (long)n * HIDD;
#pragma unroll 4
  for (int c = 0; c < HIDD; ++c) s += hp[c] * W[c];
  out[n] = 1.0f / (1.0f + __expf(-s));
}

// ---------------------------------------------------------------------------
// K10: action_logits = (pooled/counts) @ W_act + b_act  -> out[N ..]
// ---------------------------------------------------------------------------
__global__ void k_actions(const float* __restrict__ pooled, const float* __restrict__ counts,
                          const float* __restrict__ Wa, const float* __restrict__ ba,
                          float* __restrict__ out) {
  int t = blockIdx.x * blockDim.x + threadIdx.x;
  if (t >= NB * NACT) return;
  int b = t / NACT, a = t % NACT;
  float inv = 1.0f / fmaxf(counts[b], 1.0f);
  float s = ba[a];
  for (int c = 0; c < HIDD; ++c) s += pooled[b * HIDD + c] * inv * Wa[c * NACT + a];
  out[NN + t] = s;
}

// ---------------------------------------------------------------------------
extern "C" void kernel_launch(void* const* d_in, const int* in_sizes, int n_in,
                              void* d_out, int out_size, void* d_ws, size_t ws_size,
                              hipStream_t stream) {
  const float* x   = (const float*)d_in[0];
  const int*   ei  = (const int*)d_in[1];     // [2,E]: row0=src, row1=dst
  const int*   bat = (const int*)d_in[2];
  const float* Wl  = (const float*)d_in[3];
  const float* Wr  = (const float*)d_in[4];
  const float* bs  = (const float*)d_in[5];
  const float* Wg  = (const float*)d_in[6];
  const float* ats = (const float*)d_in[7];
  const float* atd = (const float*)d_in[8];
  const float* bg  = (const float*)d_in[9];
  const float* Ww  = (const float*)d_in[10];
  const float* bw  = (const float*)d_in[11];
  const float* Wt  = (const float*)d_in[12];
  const float* bt  = (const float*)d_in[13];
  const float* Wa  = (const float*)d_in[14];
  const float* ba  = (const float*)d_in[15];
  float* out = (float*)d_out;

  // Workspace layout (floats; every region is a multiple of 8 floats -> all
  // sub-buffers stay 32-byte aligned for v16bf fragment loads).
  float* ws = (float*)d_ws;
  size_t off = 0;
  float*    neigh  = ws + off; off += (size_t)NN * IND;           // N*128
  float*    deg    = ws + off; off += NN;
  float*    gbuf   = ws + off; off += (size_t)NN * NHEADS * HIDD; // N*512
  float*    a_s    = ws + off; off += (size_t)NN * NHEADS;
  float*    a_d    = ws + off; off += (size_t)NN * NHEADS;
  unsigned* mmax   = (unsigned*)(ws + off); off += (size_t)NN * NHEADS;
  float*    denom  = ws + off; off += (size_t)NN * NHEADS;
  float*    ebuf   = ws + off; off += (size_t)NE * NHEADS;        // E*4
  float*    h2     = ws + off; off += (size_t)NN * HIDD;          // N*128
  float*    pooled = ws + off; off += (size_t)NB * HIDD;
  float*    counts = ws + off; off += NB;
  __bf16*   hbf    = (__bf16*)(ws + off); off += (size_t)NN * HIDD / 2;   // bf16 h
  __bf16*   xbf    = (__bf16*)(ws + off); off += (size_t)NN * HIDD / 2;   // bf16 x
  __bf16*   mbf    = (__bf16*)(ws + off); off += (size_t)NN * HIDD / 2;   // bf16 mean
  __bf16*   Wlp    = (__bf16*)(ws + off); off += (size_t)IND * HIDD / 2;  // packed
  __bf16*   Wrp    = (__bf16*)(ws + off); off += (size_t)IND * HIDD / 2;
  __bf16*   Wgp    = (__bf16*)(ws + off); off += (size_t)HIDD * NHEADS * HIDD / 2;

  hipMemsetAsync(d_ws, 0, off * sizeof(float), stream);  // zeros == fmap-min for mmax too

  dim3 blk(256);
  // Weight pre-swizzle into WMMA fragment order (tiny, amortized over 1250 blocks)
  k_pack_w<<<(IND * HIDD + 255) / 256, blk, 0, stream>>>(Wl, HIDD, Wlp, IND * HIDD);
  k_pack_w<<<(IND * HIDD + 255) / 256, blk, 0, stream>>>(Wr, HIDD, Wrp, IND * HIDD);
  k_pack_w<<<(HIDD * NHEADS * HIDD + 255) / 256, blk, 0, stream>>>(
      Wg, NHEADS * HIDD, Wgp, HIDD * NHEADS * HIDD);

  k_sage_scatter<<<(NE * 32 + 255) / 256, blk, 0, stream>>>(x, ei, neigh, deg);
  k_cvt_inputs<<<(NN * HIDD + 255) / 256, blk, 0, stream>>>(x, neigh, deg, xbf, mbf);
  k_sage_gemm<<<NN / 16, blk, 0, stream>>>(xbf, mbf, Wlp, Wrp, bs, hbf);
  k_gat_gemm<<<dim3(NN / 16, 4), blk, 0, stream>>>(hbf, Wgp, gbuf);
  k_att_scores<<<(NN * NHEADS + 255) / 256, blk, 0, stream>>>(gbuf, ats, atd, a_s, a_d);
  k_edge_max<<<(NE * NHEADS + 255) / 256, blk, 0, stream>>>(ei, a_s, a_d, ebuf, mmax);
  k_edge_exp<<<(NE * NHEADS + 255) / 256, blk, 0, stream>>>(ei, mmax, ebuf, denom);
  k_edge_aggr<<<(NE * 32 + 255) / 256, blk, 0, stream>>>(ei, ebuf, denom, gbuf, h2);
  k_finalize<<<(NN * HIDD + 255) / 256, blk, 0, stream>>>(bg, bat, h2, pooled, counts);
  k_node_heads<<<(NN + 255) / 256, blk, 0, stream>>>(h2, Ww, bw, Wt, bt, out);
  k_actions<<<1, blk, 0, stream>>>(pooled, counts, Wa, ba, out);
}
